// KANRanker_73950746903167
// MI455X (gfx1250) — matrix operand order repaired and checked by reading, not compiled, not run on hardware
//
#include <hip/hip_runtime.h>
#include <math.h>

typedef __attribute__((ext_vector_type(16))) _Float16 v16h;
typedef __attribute__((ext_vector_type(8)))  float    v8f;

// Flattened input order assumed (setup_inputs() dict order, recursive):
// 0:x
// 1..24: ecf/ecg/cf/cg each {l1.W,l1.b,l2.W,l2.b,ln.g,ln.b}
// 25..32: mod0..3 {W,b}; 33,34: glob{W,b}; 35: query; 36,37:q; 38,39:k; 40,41:v;
// 42,43: out; 44,45: ln1; 46,47: ln2; 48,49: ff1; 50,51: ff2
// 52,53: fusion; 54..56: sp1{coeff,scale,bias}; 57,58: fc1; 59..61: sp2; 62,63: fc2; 64,65: out
struct KParams { const float* p[66]; };

static __device__ inline float frcp(float x)  { return __builtin_amdgcn_rcpf(x); }
static __device__ inline float frsq(float x)  { return __builtin_amdgcn_rsqf(x); }
static __device__ inline float siluf(float x) { return x * frcp(1.f + __expf(-x)); }

// ---------------- fragment helpers (ISA 7.12.2 layouts, wave32) ----------------

// A fragment 16x32 f16 from row-major f32 LDS tile.
// lane<16: M=lane, elems 0..7 -> K 0..7, elems 8..15 -> K 16..23
// lane>=16: M=lane-16, elems 0..7 -> K 8..15, elems 8..15 -> K 24..31
static __device__ inline v16h load_a16(const float* base, int stride, int koff, int lane) {
  int m   = lane & 15;
  int khi = (lane >> 4) * 8;
  const float* row = base + m * stride + koff + khi;
  v16h a;
#pragma unroll
  for (int j = 0; j < 4; ++j) {
    float2 lo = *(const float2*)(row + 2 * j);
    float2 hi = *(const float2*)(row + 16 + 2 * j);
    a[2 * j]     = (_Float16)lo.x; a[2 * j + 1] = (_Float16)lo.y;
    a[8 + 2 * j] = (_Float16)hi.x; a[9 + 2 * j] = (_Float16)hi.y;
  }
  return a;
}

// B fragment: prepacked in d_ws by kan_pack: lane -> N=lane&15, elem i -> K=(i&7)+8*(lane>>4)+16*(i>>3)
static __device__ inline v8f wmma_fma(v16h a, const v16h* wsfrag, int fragId, int lane, v8f c) {
  v16h b = wsfrag[fragId * 32 + lane];
  return __builtin_amdgcn_wmma_f32_16x16x32_f16(false, a, false, b, (short)0, c, false, false);
}

// D fragment (16x16 f32): VGPR r, lane -> (M = r + 8*(lane>>4), N = lane&15)
static __device__ inline void store_d(float* base, int stride, int coloff, int lane, v8f d) {
  int n  = coloff + (lane & 15);
  int rb = (lane >> 4) * 8;
#pragma unroll
  for (int r = 0; r < 8; ++r) base[(rb + r) * stride + n] = d[r];
}

static __device__ inline float red16(float v) {  // masks <16 never cross the 16-lane halves
  v += __shfl_xor(v, 1); v += __shfl_xor(v, 2);
  v += __shfl_xor(v, 4); v += __shfl_xor(v, 8);
  return v;
}

// LayerNorm over 32 features held as two 16-col D fragments (in registers).
static __device__ inline void ln_pair(v8f& a0, v8f& a1, const float* g, const float* b, int lane) {
  int n = lane & 15;
  float g0 = g[n], g1 = g[16 + n], b0 = b[n], b1 = b[16 + n];
#pragma unroll
  for (int r = 0; r < 8; ++r) {
    float s  = red16(a0[r] + a1[r]);
    float q  = red16(a0[r] * a0[r] + a1[r] * a1[r]);
    float mu = s * (1.f / 32.f);
    float vr = q * (1.f / 32.f) - mu * mu;
    float iv = frsq(vr + 1e-5f);
    a0[r] = g0 * (a0[r] - mu) * iv + b0;
    a1[r] = g1 * (a1[r] - mu) * iv + b1;
  }
}

static __device__ inline float splinef(float v, const float* coeff, float sc, float bi, int c) {
  const float step  = 7.f / 15.f;
  const float rstep = 1.f / (7.f / 15.f + 1e-8f);   // compile-time constant
  float xc = fminf(fmaxf(v, -3.5f), 3.5f);
  int i0 = (int)floorf((xc + 3.5f) * (15.f / 7.f));
  i0 = i0 < 0 ? 0 : (i0 > 14 ? 14 : i0);
  float x0 = -3.5f + (float)i0 * step;
  float t  = (xc - x0) * rstep;
  float y0 = coeff[c * 16 + i0], y1 = coeff[c * 16 + i0 + 1];
  return sc * xc + bi + y0 + t * (y1 - y0);
}

// ---------------- weight pre-pack: one block per B fragment ----------------
// frag ids: 0-7 ENC1(blockdiag 24->128) | 8-15 ENC2(e,t) | 16-23 MOD(e,t) | 24-25 GLOB
//           26-27 V | 28-29 OUT | 30-33 FF1 | 34-37 FF2(t,kc) | 38-41 FUS
//           42-49 FC1(t,kc) | 50-53 FC2(t,kc)                              (54 total)
__global__ __launch_bounds__(32) void kan_pack(KParams P, v16h* __restrict__ wsfrag) {
  int f = blockIdx.x, lane = threadIdx.x;
  int n = lane & 15;
  int kb = (lane >> 4) * 16;
  v16h frag;
#pragma unroll
  for (int i = 0; i < 16; ++i) {
    int kk = (kb >> 1) + (i & 7) + ((i >> 3) << 4);
    float val = 0.f;
    if (f < 8) {                       // combined encoder l1, block-diagonal over K
      int c = 16 * f + n; int e = c >> 5; int o = c & 31; int klo = 6 * e;
      if (kk >= klo && kk < klo + 6) val = P.p[1 + 6 * e][o * 6 + (kk - klo)];
    } else if (f < 16) { int g = f - 8;  int e = g >> 1, t = g & 1;
      val = P.p[3 + 6 * e][(16 * t + n) * 32 + kk];
    } else if (f < 24) { int g = f - 16; int e = g >> 1, t = g & 1;
      val = P.p[25 + 2 * e][(16 * t + n) * 32 + kk];
    } else if (f < 26) { int t = f - 24; if (kk < 17) val = P.p[33][(16 * t + n) * 17 + kk];
    } else if (f < 28) { int t = f - 26; val = P.p[40][(16 * t + n) * 32 + kk];
    } else if (f < 30) { int t = f - 28; val = P.p[42][(16 * t + n) * 32 + kk];
    } else if (f < 34) { int t = f - 30; val = P.p[48][(16 * t + n) * 32 + kk];
    } else if (f < 38) { int g = f - 34; int t = g >> 1, kc = g & 1;
      val = P.p[50][(16 * t + n) * 64 + 32 * kc + kk];
    } else if (f < 42) { int t = f - 38; val = P.p[52][(16 * t + n) * 32 + kk];
    } else if (f < 50) { int g = f - 42; int t = g >> 1, kc = g & 1;
      val = P.p[57][(16 * t + n) * 64 + 32 * kc + kk];
    } else {             int g = f - 50; int t = g >> 1, kc = g & 1;
      val = P.p[62][(16 * t + n) * 64 + 32 * kc + kk];
    }
    frag[i] = (_Float16)val;
  }
  wsfrag[f * 32 + lane] = frag;
}

// ---------------- fused main kernel: one wave = one 16-row tile ----------------
#define WPB 2
__global__ __launch_bounds__(32 * WPB) void kan_main(KParams P,
                                                     const v16h* __restrict__ wsfrag,
                                                     float* __restrict__ out, int nrows) {
  __shared__ __align__(16) float XT[WPB][16 * 64];    // raw x tile (cols 41..63 zeroed)
  __shared__ __align__(16) float BA[WPB][16 * 168];   // H1/E -> VH -> ATT/FF -> Hsp
  __shared__ __align__(16) float BB[WPB][16 * 168];   // KV -> AV -> F -> H2
  __shared__ __align__(16) float MS[WPB][512];        // wq_eff[4][32], cb[4], softmax w[16][4][5]

  const int lane = threadIdx.x & 31;
  const int w    = threadIdx.x >> 5;
  float* xt = &XT[w][0]; float* ba = &BA[w][0]; float* bb = &BB[w][0]; float* ms = &MS[w][0];
  const float* x = P.p[0];
  const float* query = P.p[35];

  // ---- constant folded-query precompute (query is row-independent) ----
  {
    const float* Wq = P.p[36]; const float* qb = P.p[37];
    const float* Wk = P.p[38]; const float* kb = P.p[39];
    float qh = qb[lane];
#pragma unroll 1
    for (int i = 0; i < 32; ++i) qh += query[i] * Wq[lane * 32 + i];
    const float is8 = 0.35355339059327373f;  // 1/sqrt(HD)
#pragma unroll 1
    for (int h = 0; h < 4; ++h) {
      float wqe = 0.f, cbh = 0.f;
#pragma unroll
      for (int d = 0; d < 8; ++d) {
        float qd = __shfl(qh, h * 8 + d);
        wqe += qd * Wk[(h * 8 + d) * 32 + lane];
        cbh += qd * kb[h * 8 + d];
      }
      ms[h * 32 + lane] = wqe * is8;
      if (lane == 0) ms[128 + h] = cbh * is8;
    }
  }

  const int tiles = nrows >> 4;
  const int gw = blockIdx.x * WPB + w;
  const int tw = gridDim.x * WPB;
#pragma unroll 1
  for (int tile = gw; tile < tiles; tile += tw) {
    const int r0 = tile * 16;

    // ---- prefetch next tile's rows while this tile computes ----
    {
      int nxt = tile + tw;
      if (nxt < tiles) {
        const char* pf = (const char*)(x + (long)nxt * 16 * 41);
        __builtin_prefetch(pf + lane * 128, 0, 1);  // global_prefetch_b8, ~4KB span
      }
    }

    // ---- stage x tile (no integer division: fixed columns per row) ----
#pragma unroll 1
    for (int r = 0; r < 16; ++r) {
      const float* xr = x + (long)(r0 + r) * 41;
      float* dst = xt + r * 64;
      dst[lane] = xr[lane];
      if (lane < 9)  dst[32 + lane] = xr[32 + lane];
      if (lane < 23) dst[41 + lane] = 0.f;
    }

    // ---- encoders l1 (blockdiag 24->128) + silu -> ba[0..127] ----
    {
      v16h a = load_a16(xt, 64, 0, lane);
#pragma unroll 1
      for (int t = 0; t < 8; ++t) {
        v8f acc = {};
        acc = wmma_fma(a, wsfrag, t, lane, acc);
        int c = 16 * t + (lane & 15);
        float bias = P.p[2 + 6 * (c >> 5)][c & 31];
#pragma unroll
        for (int r = 0; r < 8; ++r) acc[r] = siluf(acc[r] + bias);
        store_d(ba, 168, 16 * t, lane, acc);
      }
    }
    // ---- encoders l2 + per-encoder LN (in registers) -> E in place ----
#pragma unroll 1
    for (int e = 0; e < 4; ++e) {
      v16h a = load_a16(ba, 168, 32 * e, lane);
      v8f a0 = {}, a1 = {};
      a0 = wmma_fma(a, wsfrag, 8 + 2 * e, lane, a0);
      a1 = wmma_fma(a, wsfrag, 9 + 2 * e, lane, a1);
      const float* l2b = P.p[4 + 6 * e];
      int n = lane & 15;
      float b0 = l2b[n], b1 = l2b[16 + n];
#pragma unroll
      for (int r = 0; r < 8; ++r) { a0[r] += b0; a1[r] += b1; }
      ln_pair(a0, a1, P.p[5 + 6 * e], P.p[6 + 6 * e], lane);
      store_d(ba, 168, 32 * e, lane, a0);
      store_d(ba, 168, 32 * e + 16, lane, a1);
    }
    // ---- tokens 0..3 = mod_e(E_e) -> bb[32e..], token4 = glob(gf) -> bb[128..] ----
#pragma unroll 1
    for (int e = 0; e < 4; ++e) {
      v16h a = load_a16(ba, 168, 32 * e, lane);
      v8f a0 = {}, a1 = {};
      a0 = wmma_fma(a, wsfrag, 16 + 2 * e, lane, a0);
      a1 = wmma_fma(a, wsfrag, 17 + 2 * e, lane, a1);
      const float* mb = P.p[26 + 2 * e];
      int n = lane & 15;
#pragma unroll
      for (int r = 0; r < 8; ++r) { a0[r] += mb[n]; a1[r] += mb[16 + n]; }
      store_d(bb, 168, 32 * e, lane, a0);
      store_d(bb, 168, 32 * e + 16, lane, a1);
    }
    {
      v16h a = load_a16(xt, 64, 24, lane);   // gf, K-pad cols were zeroed
      v8f a0 = {}, a1 = {};
      a0 = wmma_fma(a, wsfrag, 24, lane, a0);
      a1 = wmma_fma(a, wsfrag, 25, lane, a1);
      const float* gb = P.p[34];
      int n = lane & 15;
#pragma unroll
      for (int r = 0; r < 8; ++r) { a0[r] += gb[n]; a1[r] += gb[16 + n]; }
      store_d(bb, 168, 128, lane, a0);
      store_d(bb, 168, 144, lane, a1);
    }
    // ---- V projection of all 5 tokens -> ba[32*tok..] (H1/E dead) ----
#pragma unroll 1
    for (int tok = 0; tok < 5; ++tok) {
      v16h a = load_a16(bb, 168, 32 * tok, lane);
      v8f a0 = {}, a1 = {};
      a0 = wmma_fma(a, wsfrag, 26, lane, a0);
      a1 = wmma_fma(a, wsfrag, 27, lane, a1);
      const float* vb = P.p[41];
      int n = lane & 15;
#pragma unroll
      for (int r = 0; r < 8; ++r) { a0[r] += vb[n]; a1[r] += vb[16 + n]; }
      store_d(ba, 168, 32 * tok, lane, a0);
      store_d(ba, 168, 32 * tok + 16, lane, a1);
    }
    // ---- scores + softmax: lane -> (row = lane&15, 2 heads) ----
    {
      int row = lane & 15, hb = (lane >> 4) * 2;
#pragma unroll 1
      for (int hh = 0; hh < 2; ++hh) {
        int h = hb + hh;
        float s[5]; float mx = -1e30f;
        const float* wqh = ms + h * 32;
        float cbh = ms[128 + h];
#pragma unroll
        for (int t = 0; t < 5; ++t) {
          const float* kv = bb + row * 168 + 32 * t;
          float acc = cbh;
#pragma unroll 1
          for (int e2 = 0; e2 < 32; ++e2) acc += wqh[e2] * kv[e2];
          s[t] = acc; mx = fmaxf(mx, acc);
        }
        float sum = 0.f;
#pragma unroll
        for (int t = 0; t < 5; ++t) { s[t] = __expf(s[t] - mx); sum += s[t]; }
        float iv = frcp(sum);
#pragma unroll
        for (int t = 0; t < 5; ++t) ms[136 + (row * 4 + h) * 5 + t] = s[t] * iv;
      }
    }
    // ---- attention value mix -> AV in bb[0..31] (KV dead) ----
    {
      int c = lane, hc = lane >> 3;
#pragma unroll 1
      for (int row = 0; row < 16; ++row) {
        float acc = 0.f;
#pragma unroll
        for (int t = 0; t < 5; ++t)
          acc += ms[136 + (row * 4 + hc) * 5 + t] * ba[row * 168 + 32 * t + c];
        bb[row * 168 + c] = acc;
      }
    }
    // ---- out-proj + query residual + LN1 -> ATT in ba[0..31] ----
    {
      v16h a = load_a16(bb, 168, 0, lane);
      v8f a0 = {}, a1 = {};
      a0 = wmma_fma(a, wsfrag, 28, lane, a0);
      a1 = wmma_fma(a, wsfrag, 29, lane, a1);
      const float* ob = P.p[43];
      int n = lane & 15;
      float d0 = ob[n] + query[n], d1 = ob[16 + n] + query[16 + n];
#pragma unroll
      for (int r = 0; r < 8; ++r) { a0[r] += d0; a1[r] += d1; }
      ln_pair(a0, a1, P.p[44], P.p[45], lane);
      store_d(ba, 168, 0, lane, a0);
      store_d(ba, 168, 16, lane, a1);
    }
    // ---- ff1 + silu -> ba[32..95] ----
    {
      v16h a = load_a16(ba, 168, 0, lane);
      const float* fb = P.p[49];
#pragma unroll 1
      for (int t = 0; t < 4; ++t) {
        v8f acc = {};
        acc = wmma_fma(a, wsfrag, 30 + t, lane, acc);
        float bias = fb[16 * t + (lane & 15)];
#pragma unroll
        for (int r = 0; r < 8; ++r) acc[r] = siluf(acc[r] + bias);
        store_d(ba, 168, 32 + 16 * t, lane, acc);
      }
    }
    // ---- ff2 (K=64) + ATT residual + LN2 -> F in bb[0..31] ----
    {
      v16h k0 = load_a16(ba, 168, 32, lane);
      v16h k1 = load_a16(ba, 168, 64, lane);
      v8f c0 = {}, c1 = {};
      c0 = wmma_fma(k0, wsfrag, 34, lane, c0);
      c0 = wmma_fma(k1, wsfrag, 35, lane, c0);
      c1 = wmma_fma(k0, wsfrag, 36, lane, c1);
      c1 = wmma_fma(k1, wsfrag, 37, lane, c1);
      const float* fb = P.p[51];
      int n = lane & 15, rb = (lane >> 4) * 8;
#pragma unroll
      for (int r = 0; r < 8; ++r) {
        c0[r] += fb[n]      + ba[(rb + r) * 168 + n];
        c1[r] += fb[16 + n] + ba[(rb + r) * 168 + 16 + n];
      }
      ln_pair(c0, c1, P.p[46], P.p[47], lane);
      store_d(bb, 168, 0, lane, c0);
      store_d(bb, 168, 16, lane, c1);
    }
    // ---- fusion + spline1 -> ba[0..63] ----
    {
      v16h a = load_a16(bb, 168, 0, lane);
      const float* fub = P.p[53]; const float* co = P.p[54];
      const float* sc  = P.p[55]; const float* bi = P.p[56];
#pragma unroll 1
      for (int t = 0; t < 4; ++t) {
        v8f acc = {};
        acc = wmma_fma(a, wsfrag, 38 + t, lane, acc);
        int c = 16 * t + (lane & 15);
        float bs = fub[c], s1 = sc[c], b1 = bi[c];
#pragma unroll
        for (int r = 0; r < 8; ++r) acc[r] = splinef(acc[r] + bs, co, s1, b1, c);
        store_d(ba, 168, 16 * t, lane, acc);
      }
    }
    // ---- fc1 (K=64) + silu + spline2 -> bb[0..63] ----
    {
      v16h k0 = load_a16(ba, 168, 0, lane);
      v16h k1 = load_a16(ba, 168, 32, lane);
      const float* fb = P.p[58]; const float* co = P.p[59];
      const float* sc = P.p[60]; const float* bi = P.p[61];
#pragma unroll 1
      for (int t = 0; t < 4; ++t) {
        v8f acc = {};
        acc = wmma_fma(k0, wsfrag, 42 + 2 * t, lane, acc);
        acc = wmma_fma(k1, wsfrag, 43 + 2 * t, lane, acc);
        int c = 16 * t + (lane & 15);
        float bs = fb[c], s2 = sc[c], b2 = bi[c];
#pragma unroll
        for (int r = 0; r < 8; ++r) acc[r] = splinef(siluf(acc[r] + bs), co, s2, b2, c);
        store_d(bb, 168, 16 * t, lane, acc);
      }
    }
    // ---- fc2 (K=64) + silu, final 32->1 dot in registers ----
    {
      v16h k0 = load_a16(bb, 168, 0, lane);
      v16h k1 = load_a16(bb, 168, 32, lane);
      v8f c0 = {}, c1 = {};
      c0 = wmma_fma(k0, wsfrag, 50, lane, c0);
      c0 = wmma_fma(k1, wsfrag, 51, lane, c0);
      c1 = wmma_fma(k0, wsfrag, 52, lane, c1);
      c1 = wmma_fma(k1, wsfrag, 53, lane, c1);
      int n = lane & 15;
      const float* f2b = P.p[63]; const float* ow = P.p[64];
      float ob2 = P.p[65][0];
      float w0 = ow[n], w1 = ow[16 + n], b0 = f2b[n], b1 = f2b[16 + n];
      float s[8];
#pragma unroll
      for (int r = 0; r < 8; ++r)
        s[r] = red16(siluf(c0[r] + b0) * w0 + siluf(c1[r] + b1) * w1);
      if (lane == 0) {
#pragma unroll
        for (int r = 0; r < 8; ++r) out[r0 + r] = s[r] + ob2;
      }
      if (lane == 16) {
#pragma unroll
        for (int r = 0; r < 8; ++r) out[r0 + 8 + r] = s[r] + ob2;
      }
    }
  }
}

extern "C" void kernel_launch(void* const* d_in, const int* in_sizes, int n_in,
                              void* d_out, int out_size, void* d_ws, size_t ws_size,
                              hipStream_t stream) {
  (void)n_in; (void)out_size; (void)ws_size;
  KParams P;
  for (int i = 0; i < 66; ++i) P.p[i] = (const float*)d_in[i];
  const int nrows = in_sizes[0] / 41;

  // 1) pre-swizzle all weights into WMMA B fragments (54 frags x 1KB in d_ws)
  kan_pack<<<54, 32, 0, stream>>>(P, (v16h*)d_ws);
  // 2) fused forward pass: 8192 waves grid-striding 65536 row-tiles
  kan_main<<<4096, 32 * WPB, 0, stream>>>(P, (const v16h*)d_ws, (float*)d_out, nrows);
}
